// DCCA_AM_30073361007252
// MI455X (gfx1250) — matrix operand order repaired and checked by reading, not compiled, not run on hardware
//
#include <hip/hip_runtime.h>
#include <cstdint>
#include <cstddef>

// ---------------------------------------------------------------------------
// DCCA with attention fusion, MI455X (gfx1250, wave32, WMMA).
// All large GEMMs use v_wmma_f32_16x16x32_f16 (f32 accumulate).
// gemm_sig: double-buffered LDS, N-tile-major grid for L2 reuse of X.
// All big-GEMM epilogues stage the 128x128 tile in LDS for coalesced b128
// stores. gemm_dh uses gfx1250 async global->LDS copies (ASYNCcnt).
// Eigen-chain (2x eigh(128) + SVD(128) via eigh(T^T T)) runs as one-workgroup
// parallel Jacobi with A,V resident in LDS (132KB dynamic LDS per WGP).
// ---------------------------------------------------------------------------

typedef __attribute__((ext_vector_type(16))) _Float16 v16h;
typedef __attribute__((ext_vector_type(8)))  float    v8f;
typedef __attribute__((ext_vector_type(4)))  float    f32x4;

union AFrag {
    v16h v;
    f32x4 f4[2];
    _Float16 h[16];
};

#define BN_ROWS 65536
#define LDSP 40   // LDS row pitch in halves (80B): keeps 16B alignment for b128 ops

// Flat address of a __shared__ object: high dword = LDS aperture, low dword =
// LDS byte offset (ISA 10.2).
__device__ inline uint32_t dcca_lds_off(const void* p) {
    return (uint32_t)(uintptr_t)p;
}
// gfx1250 async global->LDS 16B copy (tracked by ASYNCcnt).
__device__ inline void dcca_async_b128(uint32_t lds_byte_off, const void* gptr) {
    asm volatile("global_load_async_to_lds_b128 %0, %1, off"
                 :: "v"(lds_byte_off), "v"(gptr) : "memory");
}

// ------------------------------- utility kernels ---------------------------

__global__ void dcca_cvt_f16(const float* __restrict__ src, _Float16* __restrict__ dst,
                             int n, float scale) {
    for (int i = blockIdx.x * blockDim.x + threadIdx.x; i < n; i += gridDim.x * blockDim.x)
        dst[i] = (_Float16)(src[i] * scale);
}

__global__ void dcca_zero_f32(float* __restrict__ p, int n) {
    for (int i = blockIdx.x * blockDim.x + threadIdx.x; i < n; i += gridDim.x * blockDim.x)
        p[i] = 0.f;
}

__global__ void dcca_bn_finalize(const float* __restrict__ sum, const float* __restrict__ sumsq,
                                 float* __restrict__ mean, float* __restrict__ rstd, int n) {
    int i = blockIdx.x * blockDim.x + threadIdx.x;
    if (i < n) {
        float m = sum[i] * (1.f / (float)BN_ROWS);
        float v = sumsq[i] * (1.f / (float)BN_ROWS) - m * m;
        mean[i] = m;
        rstd[i] = rsqrtf(v + 1e-5f);
    }
}

// out1c_f16[b,i] = (f16)(out1[b,i] - colsum[i]/B)
__global__ void dcca_center_cvt(const float* __restrict__ O, const float* __restrict__ colsum,
                                _Float16* __restrict__ dst, int total) {
    for (int i = blockIdx.x * blockDim.x + threadIdx.x; i < total; i += gridDim.x * blockDim.x) {
        int col = i & 127;
        dst[i] = (_Float16)(O[i] - colsum[col] * (1.f / (float)BN_ROWS));
    }
}

// S = S/(m-1), +ridge on diagonal for S11 (mat 0) and S22 (mat 2)
__global__ void dcca_cov_finalize(float* __restrict__ S) {
    int idx = blockIdx.x * blockDim.x + threadIdx.x;
    if (idx < 3 * 16384) {
        int mat = idx / 16384;
        int r = idx & 16383;
        int i = r >> 7, j = r & 127;
        float v = S[idx] * (1.f / 65535.f);
        if (mat != 1 && i == j) v += 1e-3f;
        S[idx] = v;
    }
}

// -------------------------- encoder layer-1 GEMM ---------------------------
// H = sigmoid(X[M,K] @ W[N,K]^T + b); Hout f16 [M,N]; column sum/sumsq atomics.
// grid = (N/128, M/128): N fastest so both N-tiles of one M-tile share X in L2.
__global__ __launch_bounds__(256) void dcca_gemm_sig(
    const float* __restrict__ X, const _Float16* __restrict__ W,
    const float* __restrict__ bias, _Float16* __restrict__ Hout,
    float* __restrict__ colSum, float* __restrict__ colSumSq,
    int N, int K) {
    __shared__ _Float16 As[2][128][LDSP];
    __shared__ _Float16 Bs[2][128][LDSP];
    __shared__ _Float16 Cs[128][136];
    int tid = threadIdx.x;
    int n0 = blockIdx.x * 128, m0 = blockIdx.y * 128;
    int lr = tid >> 1, lc = (tid & 1) * 16;
    int wid = tid >> 5, lane = tid & 31;
    int wm = wid >> 1, wn = wid & 1;
    int g = lane >> 4, m16 = lane & 15;

    v8f acc[2][4];
    for (int mi = 0; mi < 2; ++mi)
        for (int ni = 0; ni < 4; ++ni)
            for (int e = 0; e < 8; ++e) acc[mi][ni][e] = 0.f;

    AFrag ra;        // staged X tile (converted to f16)
    f32x4 rb0, rb1;  // staged W tile
    auto fetch = [&](int k0) {
        const f32x4* srcA = (const f32x4*)(X + (size_t)(m0 + lr) * K + k0 + lc);
#pragma unroll
        for (int q2 = 0; q2 < 4; ++q2) {
            f32x4 a = srcA[q2];
#pragma unroll
            for (int j = 0; j < 4; ++j) ra.h[q2 * 4 + j] = (_Float16)a[j];
        }
        const f32x4* srcB = (const f32x4*)(W + (size_t)(n0 + lr) * K + k0 + lc);
        rb0 = srcB[0]; rb1 = srcB[1];
    };
    auto commit = [&](int buf) {
        _Float16* da = &As[buf][lr][lc];
        *(f32x4*)da = ra.f4[0];
        *(f32x4*)(da + 8) = ra.f4[1];
        _Float16* db = &Bs[buf][lr][lc];
        *(f32x4*)db = rb0;
        *(f32x4*)(db + 8) = rb1;
    };

    fetch(0);
    commit(0);
    __syncthreads();
    int nk = K >> 5;
    for (int ks = 0; ks < nk; ++ks) {
        int cur = ks & 1;
        if (ks + 1 < nk) fetch((ks + 1) * 32);
        AFrag af[2], bf[4];
#pragma unroll
        for (int mi = 0; mi < 2; ++mi) {
            const _Float16* rp = &As[cur][wm * 32 + mi * 16 + m16][0];
            af[mi].f4[0] = *(const f32x4*)(rp + g * 8);
            af[mi].f4[1] = *(const f32x4*)(rp + 16 + g * 8);
        }
#pragma unroll
        for (int ni = 0; ni < 4; ++ni) {
            const _Float16* rp = &Bs[cur][wn * 64 + ni * 16 + m16][0];
            bf[ni].f4[0] = *(const f32x4*)(rp + g * 8);
            bf[ni].f4[1] = *(const f32x4*)(rp + 16 + g * 8);
        }
#pragma unroll
        for (int mi = 0; mi < 2; ++mi)
#pragma unroll
            for (int ni = 0; ni < 4; ++ni)
                acc[mi][ni] = __builtin_amdgcn_wmma_f32_16x16x32_f16(
                    false, af[mi].v, false, bf[ni].v, (short)0, acc[mi][ni], false, false);
        if (ks + 1 < nk) commit(cur ^ 1);
        __syncthreads();
    }
    // epilogue: sigmoid + BN stats, stage f16 tile in LDS, coalesced store
#pragma unroll
    for (int ni = 0; ni < 4; ++ni) {
        int col = wn * 64 + ni * 16 + m16;
        float bv = bias[n0 + col];
        float s = 0.f, ss = 0.f;
#pragma unroll
        for (int mi = 0; mi < 2; ++mi)
#pragma unroll
            for (int r = 0; r < 8; ++r) {
                int trow = wm * 32 + mi * 16 + g * 8 + r;
                float v = acc[mi][ni][r] + bv;
                v = 1.f / (1.f + __expf(-v));
                Cs[trow][col] = (_Float16)v;
                s += v; ss += v * v;
            }
        atomicAdd(&colSum[n0 + col], s);
        atomicAdd(&colSumSq[n0 + col], ss);
    }
    __syncthreads();
    {
        int row = tid >> 1, seg = (tid & 1) * 64;
        const f32x4* sp = (const f32x4*)&Cs[row][seg];
        f32x4* dp = (f32x4*)(Hout + (size_t)(m0 + row) * N + n0 + seg);
#pragma unroll
        for (int q2 = 0; q2 < 8; ++q2) dp[q2] = sp[q2];
    }
}

// -------------------------- encoder layer-2 GEMM ---------------------------
// Out = ((H - mean)*rstd) @ W1[128,K]^T + b1 ; f32 out; column-sum atomics.
__global__ __launch_bounds__(256) void dcca_gemm_bn(
    const _Float16* __restrict__ Hbuf, const float* __restrict__ mean,
    const float* __restrict__ rstd, const _Float16* __restrict__ W,
    const float* __restrict__ bias, float* __restrict__ Out,
    float* __restrict__ colSum, int K) {
    __shared__ _Float16 As[128][LDSP];
    __shared__ _Float16 Bs[128][LDSP];
    __shared__ float Cf[128][132];
    int tid = threadIdx.x;
    int m0 = blockIdx.x * 128;
    int lr = tid >> 1, lc = (tid & 1) * 16;
    int wid = tid >> 5, lane = tid & 31;
    int wm = wid >> 1, wn = wid & 1;
    int g = lane >> 4, m16 = lane & 15;

    v8f acc[2][4];
    for (int mi = 0; mi < 2; ++mi)
        for (int ni = 0; ni < 4; ++ni)
            for (int e = 0; e < 8; ++e) acc[mi][ni][e] = 0.f;

    for (int k0 = 0; k0 < K; k0 += 32) {
        {   // BN-normalized A tile
            const f32x4* src = (const f32x4*)(Hbuf + (size_t)(m0 + lr) * K + k0 + lc);
            AFrag in, t;
            in.f4[0] = src[0]; in.f4[1] = src[1];
#pragma unroll
            for (int j = 0; j < 16; ++j)
                t.h[j] = (_Float16)(((float)in.h[j] - mean[k0 + lc + j]) * rstd[k0 + lc + j]);
            _Float16* dst = &As[lr][lc];
            *(f32x4*)dst = t.f4[0];
            *(f32x4*)(dst + 8) = t.f4[1];
        }
        {   // W1 tile (rows lr = n index 0..127)
            const f32x4* src = (const f32x4*)(W + (size_t)lr * K + k0 + lc);
            f32x4 b0 = src[0], b1 = src[1];
            _Float16* dst = &Bs[lr][lc];
            *(f32x4*)dst = b0;
            *(f32x4*)(dst + 8) = b1;
        }
        __syncthreads();
        AFrag af[2], bf[4];
#pragma unroll
        for (int mi = 0; mi < 2; ++mi) {
            const _Float16* rp = &As[wm * 32 + mi * 16 + m16][0];
            af[mi].f4[0] = *(const f32x4*)(rp + g * 8);
            af[mi].f4[1] = *(const f32x4*)(rp + 16 + g * 8);
        }
#pragma unroll
        for (int ni = 0; ni < 4; ++ni) {
            const _Float16* rp = &Bs[wn * 64 + ni * 16 + m16][0];
            bf[ni].f4[0] = *(const f32x4*)(rp + g * 8);
            bf[ni].f4[1] = *(const f32x4*)(rp + 16 + g * 8);
        }
#pragma unroll
        for (int mi = 0; mi < 2; ++mi)
#pragma unroll
            for (int ni = 0; ni < 4; ++ni)
                acc[mi][ni] = __builtin_amdgcn_wmma_f32_16x16x32_f16(
                    false, af[mi].v, false, bf[ni].v, (short)0, acc[mi][ni], false, false);
        __syncthreads();
    }
#pragma unroll
    for (int ni = 0; ni < 4; ++ni) {
        int col = wn * 64 + ni * 16 + m16;
        float bv = bias[col];
        float s = 0.f;
#pragma unroll
        for (int mi = 0; mi < 2; ++mi)
#pragma unroll
            for (int r = 0; r < 8; ++r) {
                int trow = wm * 32 + mi * 16 + g * 8 + r;
                float v = acc[mi][ni][r] + bv;
                Cf[trow][col] = v;
                s += v;
            }
        atomicAdd(&colSum[col], s);
    }
    __syncthreads();
    {
        int row = tid >> 1, seg = (tid & 1) * 64;
        const f32x4* sp = (const f32x4*)&Cf[row][seg];
        f32x4* dp = (f32x4*)(Out + (size_t)(m0 + row) * 128 + seg);
#pragma unroll
        for (int q2 = 0; q2 < 16; ++q2) dp[q2] = sp[q2];
    }
}

// ----------------------- covariance GEMMs (split-K) ------------------------
// S = A^T @ B over K=65536 batch; grid (64 K-splits, 3 matrices); f32 atomics.
__global__ __launch_bounds__(256) void dcca_cov(
    const _Float16* __restrict__ O1, const _Float16* __restrict__ O2,
    float* __restrict__ Sbase) {
    __shared__ _Float16 LA[128][LDSP];
    __shared__ _Float16 LB[128][LDSP];
    int mat = blockIdx.y;  // 0:S11 1:S12 2:S22
    const _Float16* Ap = (mat == 2) ? O2 : O1;
    const _Float16* Bp = (mat == 0) ? O1 : O2;
    float* Sout = Sbase + (size_t)mat * 16384;
    int tid = threadIdx.x;
    int wid = tid >> 5, lane = tid & 31;
    int wm = wid >> 1, wn = wid & 1;
    int g = lane >> 4, m16 = lane & 15;
    int bload = tid >> 3;           // 0..31 batch row in chunk
    int cload = (tid & 7) * 16;     // 0..112 feature column

    v8f acc[2][4];
    for (int mi = 0; mi < 2; ++mi)
        for (int ni = 0; ni < 4; ++ni)
            for (int e = 0; e < 8; ++e) acc[mi][ni][e] = 0.f;

    int kBase = blockIdx.x * 1024;
    for (int k0 = kBase; k0 < kBase + 1024; k0 += 32) {
        AFrag ta, tb;
        const f32x4* sa = (const f32x4*)(Ap + (size_t)(k0 + bload) * 128 + cload);
        ta.f4[0] = sa[0]; ta.f4[1] = sa[1];
        const f32x4* sb = (const f32x4*)(Bp + (size_t)(k0 + bload) * 128 + cload);
        tb.f4[0] = sb[0]; tb.f4[1] = sb[1];
#pragma unroll
        for (int j = 0; j < 16; ++j) {   // transpose into LDS: L[i][b]
            LA[cload + j][bload] = ta.h[j];
            LB[cload + j][bload] = tb.h[j];
        }
        __syncthreads();
        AFrag af[2], bf[4];
#pragma unroll
        for (int mi = 0; mi < 2; ++mi) {
            const _Float16* rp = &LA[wm * 32 + mi * 16 + m16][0];
            af[mi].f4[0] = *(const f32x4*)(rp + g * 8);
            af[mi].f4[1] = *(const f32x4*)(rp + 16 + g * 8);
        }
#pragma unroll
        for (int ni = 0; ni < 4; ++ni) {
            const _Float16* rp = &LB[wn * 64 + ni * 16 + m16][0];
            bf[ni].f4[0] = *(const f32x4*)(rp + g * 8);
            bf[ni].f4[1] = *(const f32x4*)(rp + 16 + g * 8);
        }
#pragma unroll
        for (int mi = 0; mi < 2; ++mi)
#pragma unroll
            for (int ni = 0; ni < 4; ++ni)
                acc[mi][ni] = __builtin_amdgcn_wmma_f32_16x16x32_f16(
                    false, af[mi].v, false, bf[ni].v, (short)0, acc[mi][ni], false, false);
        __syncthreads();
    }
#pragma unroll
    for (int ni = 0; ni < 4; ++ni) {
        int j = wn * 64 + ni * 16 + m16;
#pragma unroll
        for (int mi = 0; mi < 2; ++mi)
#pragma unroll
            for (int r = 0; r < 8; ++r) {
                int i = wm * 32 + mi * 16 + g * 8 + r;
                atomicAdd(&Sout[i * 128 + j], acc[mi][ni][r]);
            }
    }
}

// ---------------------- two-term CCA backward GEMMs ------------------------
// dH = (A1 @ WA^T + A2 @ WB^T) / (m-1); K=N=128.
// Tiles are straight copies -> gfx1250 async global->LDS loads (ASYNCcnt).
__global__ __launch_bounds__(256) void dcca_gemm_dh(
    const _Float16* __restrict__ O1, const _Float16* __restrict__ O2,
    const _Float16* __restrict__ Wd11, const _Float16* __restrict__ Wd12,
    const _Float16* __restrict__ Wd22, float* __restrict__ dH1, float* __restrict__ dH2) {
    __shared__ _Float16 As[128][LDSP];
    __shared__ _Float16 Bs[128][LDSP];
    __shared__ float Cf[128][132];
    int which = blockIdx.y;
    const _Float16* A1 = which ? O2 : O1;
    const _Float16* WA = which ? Wd22 : Wd11;
    const _Float16* A2 = which ? O1 : O2;
    const _Float16* WB = Wd12;
    float* dst = which ? dH2 : dH1;
    int tid = threadIdx.x;
    int m0 = blockIdx.x * 128;
    int lr = tid >> 1, lc = (tid & 1) * 16;
    int wid = tid >> 5, lane = tid & 31;
    int wm = wid >> 1, wn = wid & 1;
    int g = lane >> 4, m16 = lane & 15;

    uint32_t ldsA = dcca_lds_off(&As[lr][lc]);
    uint32_t ldsB = dcca_lds_off(&Bs[lr][lc]);

    v8f acc[2][4];
    for (int mi = 0; mi < 2; ++mi)
        for (int ni = 0; ni < 4; ++ni)
            for (int e = 0; e < 8; ++e) acc[mi][ni][e] = 0.f;

    for (int term = 0; term < 2; ++term) {
        const _Float16* Ap = term ? A2 : A1;
        const _Float16* Wp = term ? WB : WA;
        for (int k0 = 0; k0 < 128; k0 += 32) {
            {
                const char* gA = (const char*)(Ap + (size_t)(m0 + lr) * 128 + k0 + lc);
                dcca_async_b128(ldsA, gA);
                dcca_async_b128(ldsA + 16, gA + 16);
                const char* gB = (const char*)(Wp + (size_t)lr * 128 + k0 + lc);
                dcca_async_b128(ldsB, gB);
                dcca_async_b128(ldsB + 16, gB + 16);
                asm volatile("s_wait_asynccnt 0x0" ::: "memory");
            }
            __syncthreads();
            AFrag af[2], bf[4];
#pragma unroll
            for (int mi = 0; mi < 2; ++mi) {
                const _Float16* rp = &As[wm * 32 + mi * 16 + m16][0];
                af[mi].f4[0] = *(const f32x4*)(rp + g * 8);
                af[mi].f4[1] = *(const f32x4*)(rp + 16 + g * 8);
            }
#pragma unroll
            for (int ni = 0; ni < 4; ++ni) {
                const _Float16* rp = &Bs[wn * 64 + ni * 16 + m16][0];
                bf[ni].f4[0] = *(const f32x4*)(rp + g * 8);
                bf[ni].f4[1] = *(const f32x4*)(rp + 16 + g * 8);
            }
#pragma unroll
            for (int mi = 0; mi < 2; ++mi)
#pragma unroll
                for (int ni = 0; ni < 4; ++ni)
                    acc[mi][ni] = __builtin_amdgcn_wmma_f32_16x16x32_f16(
                        false, af[mi].v, false, bf[ni].v, (short)0, acc[mi][ni], false, false);
            __syncthreads();
        }
    }
    const float scale = 1.f / 65535.f;
#pragma unroll
    for (int ni = 0; ni < 4; ++ni) {
        int col = wn * 64 + ni * 16 + m16;
#pragma unroll
        for (int mi = 0; mi < 2; ++mi)
#pragma unroll
            for (int r = 0; r < 8; ++r) {
                int trow = wm * 32 + mi * 16 + g * 8 + r;
                Cf[trow][col] = acc[mi][ni][r] * scale;
            }
    }
    __syncthreads();
    {
        int row = tid >> 1, seg = (tid & 1) * 64;
        const f32x4* sp = (const f32x4*)&Cf[row][seg];
        f32x4* dp = (f32x4*)(dst + (size_t)(m0 + row) * 128 + seg);
#pragma unroll
        for (int q2 = 0; q2 < 16; ++q2) dp[q2] = sp[q2];
    }
}

// ------------------- one-workgroup parallel Jacobi eigh --------------------
// 128x128 symmetric; A,V in LDS (132KB dynamic). 12 sweeps, circle pairing.
__global__ __launch_bounds__(256) void dcca_jacobi128(
    const float* __restrict__ Ain, float* __restrict__ Vout, float* __restrict__ dOut) {
    extern __shared__ float sm[];
    float (*A)[129] = (float(*)[129])sm;
    float (*V)[129] = (float(*)[129])(sm + 128 * 129);
    float* cs = sm + 2 * 128 * 129;  // 128 floats (c,s per pair)
    int tid = threadIdx.x;
    for (int idx = tid; idx < 128 * 128; idx += 256) {
        int i = idx >> 7, j = idx & 127;
        A[i][j] = Ain[idx];
        V[i][j] = (i == j) ? 1.f : 0.f;
    }
    __syncthreads();
    int pg = tid >> 2, sub = tid & 3;
    for (int sweep = 0; sweep < 12; ++sweep) {
        for (int r = 0; r < 127; ++r) {
            int p, q;
            if (pg == 0) { p = 127; q = r % 127; }
            else { p = (r + pg) % 127; q = (r + 127 - pg) % 127; }
            if (p > q) { int t = p; p = q; q = t; }
            if (sub == 0) {
                float app = A[p][p], aqq = A[q][q], apq = A[p][q];
                float c = 1.f, s = 0.f;
                if (fabsf(apq) > 1e-20f) {
                    float tau = (aqq - app) / (2.f * apq);
                    float t = ((tau >= 0.f) ? 1.f : -1.f) / (fabsf(tau) + sqrtf(1.f + tau * tau));
                    c = rsqrtf(1.f + t * t);
                    s = t * c;
                }
                cs[pg * 2] = c; cs[pg * 2 + 1] = s;
            }
            __syncthreads();
            float c = cs[pg * 2], s = cs[pg * 2 + 1];
            // rows: A <- G^T A
            for (int j = sub * 32; j < sub * 32 + 32; ++j) {
                float apj = A[p][j], aqj = A[q][j];
                A[p][j] = c * apj - s * aqj;
                A[q][j] = s * apj + c * aqj;
            }
            __syncthreads();
            // cols: A <- A G ; V <- V G
            for (int i = sub * 32; i < sub * 32 + 32; ++i) {
                float aip = A[i][p], aiq = A[i][q];
                A[i][p] = c * aip - s * aiq;
                A[i][q] = s * aip + c * aiq;
                float vip = V[i][p], viq = V[i][q];
                V[i][p] = c * vip - s * viq;
                V[i][q] = s * vip + c * viq;
            }
            __syncthreads();
        }
    }
    for (int idx = tid; idx < 128 * 128; idx += 256) {
        int i = idx >> 7, j = idx & 127;
        Vout[idx] = V[i][j];
    }
    if (tid < 128) dOut[tid] = A[tid][tid];
}

// -------------------------- small 128x128 matmuls --------------------------
__device__ inline float dcca_kscale(int mode, float d) {
    if (mode == 0) return 1.f;
    if (mode == 1) return d > 0.f ? rsqrtf(d) : 0.f;  // d^-1/2 with eig clamp
    return sqrtf(fmaxf(d, 0.f));                      // mode 2: s = sqrt(d)
}
__device__ inline float dcca_pscale(int mode, float d) {
    if (mode == 0) return 1.f;
    float s = sqrtf(fmaxf(d, 0.f));                   // mode 1: 1/s (guarded)
    return s > 1e-12f ? 1.f / s : 0.f;
}
// C[i,j] = alpha * pf(j) * sum_k opA(i,k) * f(k) * opB(k,j)
__global__ __launch_bounds__(128) void dcca_mm128(
    const float* __restrict__ A, const float* __restrict__ Bm, float* __restrict__ C,
    int transA, int transB, const float* __restrict__ dvec, int kmode,
    const float* __restrict__ pvec, int pmode, float alpha) {
    int i = blockIdx.x, j = threadIdx.x;
    float sum = 0.f;
    for (int k = 0; k < 128; ++k) {
        float a = transA ? A[k * 128 + i] : A[i * 128 + k];
        float b = transB ? Bm[j * 128 + k] : Bm[k * 128 + j];
        float f = dcca_kscale(kmode, dvec ? dvec[k] : 0.f);
        sum += a * f * b;
    }
    float pf = dcca_pscale(pmode, pvec ? pvec[j] : 0.f);
    C[i * 128 + j] = alpha * pf * sum;
}

__global__ void dcca_corr(const float* __restrict__ T, float* __restrict__ lossOut) {
    __shared__ float red[256];
    float s = 0.f;
    for (int i = threadIdx.x; i < 16384; i += 256) { float v = T[i]; s += v * v; }
    red[threadIdx.x] = s;
    __syncthreads();
    for (int off = 128; off; off >>= 1) {
        if (threadIdx.x < (unsigned)off) red[threadIdx.x] += red[threadIdx.x + off];
        __syncthreads();
    }
    if (threadIdx.x == 0) *lossOut = -sqrtf(red[0]);   // loss = -corr
}

// --------------------- attention fusion + classifier -----------------------
__global__ __launch_bounds__(256) void dcca_fuse_cls(
    const float* __restrict__ out1, const float* __restrict__ out2,
    const float* __restrict__ att_w, const float* __restrict__ clsW,
    const float* __restrict__ clsb, float* __restrict__ fused,
    float* __restrict__ alpha, float* __restrict__ outc) {
    int warp = threadIdx.x >> 5, lane = threadIdx.x & 31;
    int row = blockIdx.x * 8 + warp;
    const float* p1 = out1 + (size_t)row * 128;
    const float* p2 = out2 + (size_t)row * 128;
    float o1[4], o2[4];
    float t1 = 0.f, t2 = 0.f;
#pragma unroll
    for (int j = 0; j < 4; ++j) {
        int i = lane + j * 32;
        o1[j] = p1[i]; o2[j] = p2[i];
        float w = att_w[i];
        t1 += o1[j] * w; t2 += o2[j] * w;
    }
    for (int off = 16; off; off >>= 1) {
        t1 += __shfl_xor(t1, off);
        t2 += __shfl_xor(t2, off);
    }
    float mx = fmaxf(t1, t2);
    float n1 = __expf(t1 - mx), n2 = __expf(t2 - mx);
    float a1 = n1 / (n1 + n2);
    a1 = fminf(fmaxf(a1, 0.f), 1.f);
    float a2 = 1.f - a1;
    float c0 = 0.f, c1 = 0.f, c2 = 0.f;
    float* pf = fused + (size_t)row * 128;
#pragma unroll
    for (int j = 0; j < 4; ++j) {
        int i = lane + j * 32;
        float f = a1 * o1[j] + a2 * o2[j];
        pf[i] = f;
        c0 += f * clsW[0 * 128 + i];
        c1 += f * clsW[1 * 128 + i];
        c2 += f * clsW[2 * 128 + i];
    }
    for (int off = 16; off; off >>= 1) {
        c0 += __shfl_xor(c0, off);
        c1 += __shfl_xor(c1, off);
        c2 += __shfl_xor(c2, off);
    }
    if (lane == 0) {
        alpha[row * 2 + 0] = a1;
        alpha[row * 2 + 1] = a2;
        outc[row * 3 + 0] = c0 + clsb[0];
        outc[row * 3 + 1] = c1 + clsb[1];
        outc[row * 3 + 2] = c2 + clsb[2];
    }
}

// ------------------------------- launcher ----------------------------------
extern "C" void kernel_launch(void* const* d_in, const int* in_sizes, int n_in,
                              void* d_out, int out_size, void* d_ws, size_t ws_size,
                              hipStream_t stream) {
    (void)in_sizes; (void)n_in; (void)out_size; (void)ws_size;
    const float* x1    = (const float*)d_in[0];
    const float* x2    = (const float*)d_in[1];
    const float* m1_W0 = (const float*)d_in[2];
    const float* m1_b0 = (const float*)d_in[3];
    const float* m1_W1 = (const float*)d_in[4];
    const float* m1_b1 = (const float*)d_in[5];
    const float* m2_W0 = (const float*)d_in[6];
    const float* m2_b0 = (const float*)d_in[7];
    const float* m2_W1 = (const float*)d_in[8];
    const float* m2_b1 = (const float*)d_in[9];
    const float* att_w = (const float*)d_in[10];
    const float* cls_W = (const float*)d_in[11];
    const float* cls_b = (const float*)d_in[12];
    float* out = (float*)d_out;

    const size_t Bn = BN_ROWS;
    size_t off_out   = 0;
    size_t off_loss  = Bn * 3;
    size_t off_out1  = off_loss + 1;
    size_t off_out2  = off_out1 + Bn * 128;
    size_t off_dH1   = off_out2 + Bn * 128;
    size_t off_dH2   = off_dH1 + Bn * 128;
    size_t off_fused = off_dH2 + Bn * 128;
    size_t off_alpha = off_fused + Bn * 128;

    uint8_t* wsp = (uint8_t*)d_ws;
    auto carve = [&](size_t bytes) -> void* {
        void* p = (void*)wsp;
        wsp += (bytes + 255) & ~(size_t)255;
        return p;
    };
    _Float16* h1   = (_Float16*)carve(Bn * 256 * 2);
    _Float16* h2   = (_Float16*)carve(Bn * 256 * 2);
    _Float16* o1c  = (_Float16*)carve(Bn * 128 * 2);
    _Float16* o2c  = (_Float16*)carve(Bn * 128 * 2);
    _Float16* w0a  = (_Float16*)carve(256 * 1024 * 2);
    _Float16* w0b  = (_Float16*)carve(256 * 512 * 2);
    _Float16* w1a  = (_Float16*)carve(128 * 256 * 2);
    _Float16* w1b  = (_Float16*)carve(128 * 256 * 2);
    _Float16* wd11 = (_Float16*)carve(128 * 128 * 2);
    _Float16* wd12 = (_Float16*)carve(128 * 128 * 2);
    _Float16* wd22 = (_Float16*)carve(128 * 128 * 2);
    // zeroed (atomic) region: bn stats + out sums + S matrices
    float* zreg = (float*)carve((size_t)(1280 + 3 * 16384) * 4);
    float* bn1sum = zreg;         float* bn1ss = zreg + 256;
    float* bn2sum = zreg + 512;   float* bn2ss = zreg + 768;
    float* osum1  = zreg + 1024;  float* osum2 = zreg + 1152;
    float* Smats  = zreg + 1280;
    float* S11 = Smats, *S12 = Smats + 16384, *S22 = Smats + 32768;
    const int nZero = 1280 + 3 * 16384;
    // non-zeroed small buffers
    float* mean1 = (float*)carve(256 * 4);  float* rstd1 = (float*)carve(256 * 4);
    float* mean2 = (float*)carve(256 * 4);  float* rstd2 = (float*)carve(256 * 4);
    float* V1 = (float*)carve(16384 * 4);   float* V2 = (float*)carve(16384 * 4);
    float* V3 = (float*)carve(16384 * 4);
    float* dg1 = (float*)carve(256 * 4);    float* dg2 = (float*)carve(256 * 4);
    float* dg3 = (float*)carve(256 * 4);
    float* inv1 = (float*)carve(16384 * 4); float* inv2 = (float*)carve(16384 * 4);
    float* Tm   = (float*)carve(16384 * 4); float* M2   = (float*)carve(16384 * 4);
    float* Um   = (float*)carve(16384 * 4); float* UVt  = (float*)carve(16384 * 4);
    float* UDUt = (float*)carve(16384 * 4); float* TMP  = (float*)carve(16384 * 4);
    float* D11  = (float*)carve(16384 * 4); float* D12  = (float*)carve(16384 * 4);
    float* D22  = (float*)carve(16384 * 4);

    dim3 blk(256);
    // 1) convert weights to f16; zero accumulation region
    dcca_cvt_f16<<<dim3(1024), blk, 0, stream>>>(m1_W0, w0a, 256 * 1024, 1.f);
    dcca_cvt_f16<<<dim3(512),  blk, 0, stream>>>(m2_W0, w0b, 256 * 512, 1.f);
    dcca_cvt_f16<<<dim3(128),  blk, 0, stream>>>(m1_W1, w1a, 128 * 256, 1.f);
    dcca_cvt_f16<<<dim3(128),  blk, 0, stream>>>(m2_W1, w1b, 128 * 256, 1.f);
    dcca_zero_f32<<<dim3((nZero + 255) / 256), blk, 0, stream>>>(zreg, nZero);
    // 2) encoder layer 1 (sigmoid + BN stats); grid (N-tiles, M-tiles)
    dcca_gemm_sig<<<dim3(2, 512), blk, 0, stream>>>(x1, w0a, m1_b0, h1, bn1sum, bn1ss, 256, 1024);
    dcca_gemm_sig<<<dim3(2, 512), blk, 0, stream>>>(x2, w0b, m2_b0, h2, bn2sum, bn2ss, 256, 512);
    dcca_bn_finalize<<<dim3(1), blk, 0, stream>>>(bn1sum, bn1ss, mean1, rstd1, 256);
    dcca_bn_finalize<<<dim3(1), blk, 0, stream>>>(bn2sum, bn2ss, mean2, rstd2, 256);
    // 3) encoder layer 2 (BN fused into A load) -> out1/out2 f32 + column sums
    dcca_gemm_bn<<<dim3(512, 1), blk, 0, stream>>>(h1, mean1, rstd1, w1a, m1_b1, out + off_out1, osum1, 256);
    dcca_gemm_bn<<<dim3(512, 1), blk, 0, stream>>>(h2, mean2, rstd2, w1b, m2_b1, out + off_out2, osum2, 256);
    // 4) center + f16 convert
    dcca_center_cvt<<<dim3(4096), blk, 0, stream>>>(out + off_out1, osum1, o1c, (int)(Bn * 128));
    dcca_center_cvt<<<dim3(4096), blk, 0, stream>>>(out + off_out2, osum2, o2c, (int)(Bn * 128));
    // 5) covariances S11,S12,S22 (split-K WMMA + atomics) and finalize
    dcca_cov<<<dim3(64, 3), blk, 0, stream>>>(o1c, o2c, Smats);
    dcca_cov_finalize<<<dim3(192), blk, 0, stream>>>(Smats);
    // 6) eigh(S11), eigh(S22)
    size_t jsm = (size_t)(2 * 128 * 129 + 128) * sizeof(float);
    dcca_jacobi128<<<dim3(1), blk, jsm, stream>>>(S11, V1, dg1);
    dcca_jacobi128<<<dim3(1), blk, jsm, stream>>>(S22, V2, dg2);
    // 7) inv1/inv2 = V diag(d^-1/2, clamped) V^T ;  T = inv1 @ S12 @ inv2
    dcca_mm128<<<dim3(128), dim3(128), 0, stream>>>(V1, V1, inv1, 0, 1, dg1, 1, nullptr, 0, 1.f);
    dcca_mm128<<<dim3(128), dim3(128), 0, stream>>>(V2, V2, inv2, 0, 1, dg2, 1, nullptr, 0, 1.f);
    dcca_mm128<<<dim3(128), dim3(128), 0, stream>>>(inv1, S12, TMP, 0, 0, nullptr, 0, nullptr, 0, 1.f);
    dcca_mm128<<<dim3(128), dim3(128), 0, stream>>>(TMP, inv2, Tm, 0, 0, nullptr, 0, nullptr, 0, 1.f);
    // 8) loss = -||T||_F ;  SVD of T via eigh(T^T T)
    dcca_mm128<<<dim3(128), dim3(128), 0, stream>>>(Tm, Tm, M2, 1, 0, nullptr, 0, nullptr, 0, 1.f);
    dcca_corr<<<dim3(1), blk, 0, stream>>>(Tm, out + off_loss);
    dcca_jacobi128<<<dim3(1), blk, jsm, stream>>>(M2, V3, dg3);
    // U = T V3 diag(1/s);  UVt = U V3^T;  UDUt = U diag(s) U^T
    dcca_mm128<<<dim3(128), dim3(128), 0, stream>>>(Tm, V3, Um, 0, 0, nullptr, 0, dg3, 1, 1.f);
    dcca_mm128<<<dim3(128), dim3(128), 0, stream>>>(Um, V3, UVt, 0, 1, nullptr, 0, nullptr, 0, 1.f);
    dcca_mm128<<<dim3(128), dim3(128), 0, stream>>>(Um, Um, UDUt, 0, 1, dg3, 2, nullptr, 0, 1.f);
    // Delta12 = inv1 UVt inv2 ; Delta11 = -0.5 inv1 UDUt inv1 ; Delta22 = -0.5 inv2 UDUt inv2
    dcca_mm128<<<dim3(128), dim3(128), 0, stream>>>(inv1, UVt, TMP, 0, 0, nullptr, 0, nullptr, 0, 1.f);
    dcca_mm128<<<dim3(128), dim3(128), 0, stream>>>(TMP, inv2, D12, 0, 0, nullptr, 0, nullptr, 0, 1.f);
    dcca_mm128<<<dim3(128), dim3(128), 0, stream>>>(inv1, UDUt, TMP, 0, 0, nullptr, 0, nullptr, 0, 1.f);
    dcca_mm128<<<dim3(128), dim3(128), 0, stream>>>(TMP, inv1, D11, 0, 0, nullptr, 0, nullptr, 0, -0.5f);
    dcca_mm128<<<dim3(128), dim3(128), 0, stream>>>(inv2, UDUt, TMP, 0, 0, nullptr, 0, nullptr, 0, 1.f);
    dcca_mm128<<<dim3(128), dim3(128), 0, stream>>>(TMP, inv2, D22, 0, 0, nullptr, 0, nullptr, 0, -0.5f);
    // 9) dH1/dH2 via two-term WMMA GEMMs (fold factor 2 into f16 weights)
    dcca_cvt_f16<<<dim3(64), blk, 0, stream>>>(D11, wd11, 16384, 2.f);
    dcca_cvt_f16<<<dim3(64), blk, 0, stream>>>(D12, wd12, 16384, 1.f);
    dcca_cvt_f16<<<dim3(64), blk, 0, stream>>>(D22, wd22, 16384, 2.f);
    dcca_gemm_dh<<<dim3(512, 2), blk, 0, stream>>>(o1c, o2c, wd11, wd12, wd22,
                                                   out + off_dH1, out + off_dH2);
    // 10) attention fusion + classifier
    dcca_fuse_cls<<<dim3(8192), blk, 0, stream>>>(out + off_out1, out + off_out2, att_w,
                                                  cls_W, cls_b, out + off_fused,
                                                  out + off_alpha, out + off_out);
}
